// FFM_89550068122155
// MI455X (gfx1250) — matrix operand order repaired and checked by reading, not compiled, not run on hardware
//
#include <hip/hip_runtime.h>

#define FEAT  2048
#define FIELD 32
#define KEMB  64
#define NCOL  2048   // FIELD*KEMB
#define NTILE 128    // NCOL/16
#define NKS   64     // FEAT/32

typedef __attribute__((ext_vector_type(16))) __bf16 v16bf;
typedef __attribute__((ext_vector_type(8)))  float  v8f;

static __device__ __forceinline__ unsigned f2bf(float f) {
  union { float f; unsigned u; } a; a.f = f;
  unsigned r = a.u + 0x7FFFu + ((a.u >> 16) & 1u);   // round-to-nearest-even
  return r >> 16;
}

// ---------------------------------------------------------------------------
// Pre-pass: convert V (f32) into bf16, pre-swizzled into the per-lane WMMA
// B-fragment layout: dword index = ((t*NKS + ks)*32 + lane)*8 + j
// where dword j of lane (hl,n) packs elements f = ks*32 + (j>>2)*16 + hl*8 + 2(j&3), +1
// of column c = 16t + n.  Total 128*64*32*8 dwords = 8 MB.
// ---------------------------------------------------------------------------
__global__ __launch_bounds__(256)
void FFM_pack_v(const float* __restrict__ v, unsigned* __restrict__ vp) {
  const int idx  = blockIdx.x * 256 + threadIdx.x;
  const int j    = idx & 7;
  const int lane = (idx >> 3) & 31;
  const int ks   = (idx >> 8) & 63;
  const int t    = idx >> 14;
  const int hl = lane >> 4, n = lane & 15;
  const int f = ks * 32 + (j >> 2) * 16 + hl * 8 + 2 * (j & 3);
  const int c = t * 16 + n;
  const float lo = v[(size_t)f * NCOL + c];
  const float hi = v[(size_t)(f + 1) * NCOL + c];
  vp[idx] = f2bf(lo) | (f2bf(hi) << 16);
}

// ---------------------------------------------------------------------------
// Main fused FFM kernel: 16 rows per block, 8 waves, WMMA bf16 GEMM with
// register-resident field-sum / square-sum reductions.
// ---------------------------------------------------------------------------
__global__ __launch_bounds__(256)
void FFM_89550068122155_kernel(const float* __restrict__ x,
                               const float* __restrict__ w,
                               const float* __restrict__ bptr,
                               const unsigned* __restrict__ vp,
                               float* __restrict__ out)
{
  __shared__ unsigned short lds_x[16][FEAT];  // X row block as bf16 bits (64 KB)
  __shared__ float lds_s[16][KEMB];           // per-row field-sum s[m][k]
  __shared__ float lds_sq[16];                // per-row sum of squares
  __shared__ float lds_lin[16];               // per-row linear part x.w
  __shared__ float lds_s2[16];                // per-row ||s||^2

  const int tid  = threadIdx.x;
  const int row0 = blockIdx.x * 16;

  for (int i = tid; i < 16 * KEMB; i += 256) ((float*)lds_s)[i] = 0.f;
  if (tid < 16) { lds_sq[tid] = 0.f; lds_lin[tid] = 0.f; lds_s2[tid] = 0.f; }

  // ---- stage X -> bf16 LDS, fuse linear part (coalesced f32 loads) ----
  for (int m = 0; m < 16; ++m) {
    const float* xr = x + (size_t)(row0 + m) * FEAT;
    float lacc = 0.f;
    #pragma unroll
    for (int j = 0; j < FEAT / 256; ++j) {
      const int f = j * 256 + tid;
      const float xv = xr[f];
      lds_x[m][f] = (unsigned short)f2bf(xv);
      lacc += xv * w[f];
    }
    atomicAdd(&lds_lin[m], lacc);
  }
  __syncthreads();

  const int wave = tid >> 5;
  const int lane = tid & 31;
  const int hl   = lane >> 4;   // lane half
  const int n    = lane & 15;   // B/C column within tile; also A-fragment row

  float sacc[8], sqacc[8];
  #pragma unroll
  for (int r = 0; r < 8; ++r) { sacc[r] = 0.f; sqacc[r] = 0.f; }

  // A fragment is piecewise contiguous: VGPR0-3 = 8 bf16 @ kk+hl*8,
  // VGPR4-7 = 8 bf16 @ kk+16+hl*8  -> two ds_load_b128.
  const unsigned short* arow = &lds_x[n][0] + hl * 8;

  for (int chunk = 0; chunk < 16; ++chunk) {
    const int t = chunk * 8 + wave;          // tile -> columns [16t, 16t+16)
    v8f acc = {0.f, 0.f, 0.f, 0.f, 0.f, 0.f, 0.f, 0.f};

    // per-lane packed B stream: 2 uint4 per kstep, stride 64 uint4/kstep
    const uint4* bp = (const uint4*)vp + ((size_t)t * NKS * 32 + lane) * 2;

    #pragma unroll 4
    for (int ks = 0; ks < NKS; ++ks) {
      union { v16bf vv; uint4 q[2]; } A, Bf;
      const unsigned short* a = arow + ks * 32;
      A.q[0] = *(const uint4*)(a);          // ds_load_b128
      A.q[1] = *(const uint4*)(a + 16);     // ds_load_b128
      Bf.q[0] = bp[(size_t)ks * 64];        // global_load_b128
      Bf.q[1] = bp[(size_t)ks * 64 + 1];    // global_load_b128
      acc = __builtin_amdgcn_wmma_f32_16x16x32_bf16(
                false, A.vv, false, Bf.vv, (short)0, acc, false, false);
    }

    // C layout: lane holds column n, VGPR r -> row m = r + 8*hl.
    // This wave's k-embedding slot is fixed: kemb = 16*(wave%4) + n.
    #pragma unroll
    for (int r = 0; r < 8; ++r) {
      const float cv = acc[r];
      sacc[r]  += cv;        // sum over fields handled by this wave
      sqacc[r] += cv * cv;   // sum of squares over all columns
    }
  }

  // ---- reduce register accumulators into LDS ----
  const int kemb = 16 * (wave & 3) + n;
  #pragma unroll
  for (int r = 0; r < 8; ++r) {
    const int m = r + 8 * hl;
    atomicAdd(&lds_s[m][kemb], sacc[r]);
    atomicAdd(&lds_sq[m], sqacc[r]);
  }
  __syncthreads();

  // ---- ||s||^2 per row ----
  {
    const int m = tid >> 4, seg = tid & 15;
    float p = 0.f;
    #pragma unroll
    for (int q = 0; q < 4; ++q) {
      const float sv = lds_s[m][seg * 4 + q];
      p += sv * sv;
    }
    atomicAdd(&lds_s2[m], p);
  }
  __syncthreads();

  // ---- sigmoid(lin + b + 0.5*(||s||^2 - sq)) ----
  if (tid < 16) {
    const float z = lds_lin[tid] + bptr[0] + 0.5f * (lds_s2[tid] - lds_sq[tid]);
    out[row0 + tid] = 1.f / (1.f + __expf(-z));
  }
}

extern "C" void kernel_launch(void* const* d_in, const int* in_sizes, int n_in,
                              void* d_out, int out_size, void* d_ws, size_t ws_size,
                              hipStream_t stream) {
  (void)in_sizes; (void)n_in; (void)ws_size; (void)out_size;
  const float* x = (const float*)d_in[0];   // [B, FEAT]
  const float* w = (const float*)d_in[1];   // [FEAT]
  const float* b = (const float*)d_in[2];   // scalar
  const float* v = (const float*)d_in[3];   // [FEAT, FIELD, K]
  float* out  = (float*)d_out;              // [B]
  unsigned* vp = (unsigned*)d_ws;           // 8 MB packed bf16 B fragments

  // Pre-pass: 128*64*32*8 dwords / 256 threads = 8192 blocks
  FFM_pack_v<<<(NTILE * NKS * 32 * 8) / 256, 256, 0, stream>>>(v, vp);

  const int B = 16384;
  FFM_89550068122155_kernel<<<B / 16, 256, 0, stream>>>(x, w, b, vp, out);
}
